// GraphNetBlock_8813272891811
// MI455X (gfx1250) — matrix operand order repaired and checked by reading, not compiled, not run on hardware
//
#include <hip/hip_runtime.h>
#include <hip/hip_bf16.h>

// ---------------------------------------------------------------------------
// GraphNet block for MI455X (gfx1250, wave32).
// bf16 WMMA (v_wmma_f32_16x16x32_bf16) for both MLP GEMMs, f32 epilogue
// (bias / ReLU / LayerNorm / residual), fused scatter-add for aggregation.
// 64 rows per block (4 M-tiles) so each weight B-fragment load is reused by
// 4 WMMAs -> 4x less L2 weight traffic than a 16-row tile.
// ---------------------------------------------------------------------------

typedef __attribute__((ext_vector_type(16))) __bf16 v16bf;
typedef __attribute__((ext_vector_type(8)))  __bf16 v8bf;
typedef __attribute__((ext_vector_type(8)))  float  v8f;

#define WMMA_BF16(a, b, c) \
  __builtin_amdgcn_wmma_f32_16x16x32_bf16(false, (a), false, (b), (short)0, (c), false, false)

// Load a 16x32 bf16 A-fragment from a row-major LDS buffer.
// ISA layout (16-bit A, 16x32): lane<16 holds K = k0+[0..7] and k0+[16..23];
// lane>=16 holds K = k0+[8..15] and k0+[24..31]; row M = lane&15.
__device__ inline v16bf load_a_frag(const __bf16* buf, int ldk, int lane, int k0) {
  const int m  = lane & 15;
  const int kb = k0 + ((lane & 16) ? 8 : 0);
  const __bf16* p = buf + m * ldk + kb;
  v8bf a0 = *(const v8bf*)(p);       // ds_load_b128
  v8bf a1 = *(const v8bf*)(p + 16);  // ds_load_b128
  v16bf a;
#pragma unroll
  for (int i = 0; i < 8; ++i) { a[i] = a0[i]; a[i + 8] = a1[i]; }
  return a;
}

// B-fragments are pre-packed contiguously per (k-tile, n-tile, lane).
__device__ inline v16bf load_b_frag(const __bf16* wp, int kt, int nt, int lane) {
  return *(const v16bf*)(wp + (((size_t)(kt * 8 + nt) * 32 + lane) << 4));
}

// Re-pack a K x 128 f32 weight matrix into bf16 WMMA B-fragment order.
// 32x16 bf16 B tile: lane<16 holds K=[0..7],[16..23]; lane>=16 K=[8..15],[24..31];
// element e of the v16bf = K = kt*32 + base + e + (e>=8 ? 8 : 0), base = (lane>=16 ? 8 : 0), column n = nt*16 + (lane&15).
__global__ void pack_weights_kernel(const float* __restrict__ src,
                                    __bf16* __restrict__ dst, int K) {
  int idx = blockIdx.x * blockDim.x + threadIdx.x;
  if (idx >= K * 128) return;
  int e    = idx & 15;
  int lane = (idx >> 4) & 31;
  int nt   = (idx >> 9) & 7;
  int kt   = idx >> 12;
  int k = kt * 32 + ((lane & 16) ? 8 : 0) + e + ((e & 8) ? 8 : 0);
  int n = nt * 16 + (lane & 15);
  dst[idx] = (__bf16)src[(size_t)k * 128 + n];
}

// ---------------------------------------------------------------------------
// Edge update: new_edge = LN(relu([snd|rcv|edge] @ W1 + b1) @ W2 + b2)
// Writes new_edge + edge_feats to output, scatter-adds new_edge into agg.
// Block: 64 edges, 256 threads (8 waves); wave w owns N-tile w, all 4 M-tiles.
// ---------------------------------------------------------------------------
__global__ __launch_bounds__(256) void edge_update_kernel(
    const float* __restrict__ node_feats, const float* __restrict__ edge_feats,
    const int* __restrict__ senders, const int* __restrict__ receivers,
    const __bf16* __restrict__ w1p, const float* __restrict__ b1,
    const __bf16* __restrict__ w2p, const float* __restrict__ b2,
    const float* __restrict__ gam, const float* __restrict__ bet,
    float* __restrict__ agg, float* __restrict__ out_edge, int nE) {
  // Region 0 (48 KB): bf16 input 64x384 during GEMM1; f32 64x128 after GEMM2.
  // Region 1 (16 KB): bf16 hidden 64x128.
  __shared__ __attribute__((aligned(32))) char smem[49152 + 16384 + 256];
  __bf16* Abuf  = (__bf16*)smem;
  float*  H2    = (float*)smem;
  __bf16* Hbuf  = (__bf16*)(smem + 49152);
  int*    recvs = (int*)(smem + 49152 + 16384);

  const int t   = threadIdx.x;
  const int row = t >> 2;   // 0..63: edge row owned for gather / LN
  const int s4  = t & 3;    // 0..3 : 32-column slice within a row
  const int e   = blockIdx.x * 64 + row;
  const int eC  = (e < nE) ? e : (nE - 1);

  const int snd = senders[eC];
  const int rcv = receivers[eC];
  if (s4 == 0) recvs[row] = rcv;

  // Gather + concat + f32->bf16 into LDS: [sender(128) | receiver(128) | edge(128)]
  {
    const float* segs[3] = { node_feats + (size_t)snd * 128,
                             node_feats + (size_t)rcv * 128,
                             edge_feats + (size_t)eC  * 128 };
#pragma unroll
    for (int sg = 0; sg < 3; ++sg) {
      const float4* sp = (const float4*)segs[sg] + s4 * 8;  // 32 floats / thread
      __bf16* dp = &Abuf[row * 384 + sg * 128 + s4 * 32];
#pragma unroll
      for (int q = 0; q < 8; ++q) {
        float4 v = sp[q];
        dp[q * 4 + 0] = (__bf16)v.x; dp[q * 4 + 1] = (__bf16)v.y;
        dp[q * 4 + 2] = (__bf16)v.z; dp[q * 4 + 3] = (__bf16)v.w;
      }
    }
  }
  __syncthreads();

  const int lane = t & 31;
  const int nt   = t >> 5;               // wave id == N-tile id (0..7)
  const int col  = nt * 16 + (lane & 15);
  const int rb   = (lane & 16) ? 8 : 0;  // C/D rows: M = v + rb

  // GEMM1: 64x384 @ 384x128, K = 12 steps of 32; B reused across 4 M-tiles.
  v8f acc[4] = {};
#pragma unroll
  for (int kt = 0; kt < 12; ++kt) {
    v16bf b = load_b_frag(w1p, kt, nt, lane);
#pragma unroll
    for (int mt = 0; mt < 4; ++mt) {
      v16bf a = load_a_frag(Abuf + mt * 16 * 384, 384, lane, kt * 32);
      acc[mt] = WMMA_BF16(a, b, acc[mt]);
    }
  }
  {  // bias + ReLU, stage hidden as bf16 for GEMM2
    const float bb = b1[col];
#pragma unroll
    for (int mt = 0; mt < 4; ++mt)
#pragma unroll
      for (int v = 0; v < 8; ++v) {
        float h = acc[mt][v] + bb; h = h > 0.f ? h : 0.f;
        Hbuf[(mt * 16 + v + rb) * 128 + col] = (__bf16)h;
      }
  }
  __syncthreads();  // also: all waves done reading Abuf -> region 0 reusable

  // GEMM2: 64x128 @ 128x128, K = 4 steps of 32.
  v8f d[4] = {};
#pragma unroll
  for (int kt = 0; kt < 4; ++kt) {
    v16bf b = load_b_frag(w2p, kt, nt, lane);
#pragma unroll
    for (int mt = 0; mt < 4; ++mt) {
      v16bf a = load_a_frag(Hbuf + mt * 16 * 128, 128, lane, kt * 32);
      d[mt] = WMMA_BF16(a, b, d[mt]);
    }
  }
  {
    const float bb = b2[col];
#pragma unroll
    for (int mt = 0; mt < 4; ++mt)
#pragma unroll
      for (int v = 0; v < 8; ++v)
        H2[(mt * 16 + v + rb) * 128 + col] = d[mt][v] + bb;
  }
  __syncthreads();

  // LayerNorm over 128 cols per row; 4 lanes per row, xor-shuffle reduce.
  float sum = 0.f, sq = 0.f;
#pragma unroll
  for (int q = 0; q < 32; ++q) {
    const float x = H2[row * 128 + s4 * 32 + q];
    sum += x; sq += x * x;
  }
#pragma unroll
  for (int m = 2; m >= 1; m >>= 1) {
    sum += __shfl_xor(sum, m);
    sq  += __shfl_xor(sq, m);
  }
  const float mu  = sum * (1.f / 128.f);
  const float var = sq * (1.f / 128.f) - mu * mu;
  const float inv = rsqrtf(var + 1e-5f);

  if (e < nE) {
    const float* ef = edge_feats + (size_t)e * 128;
    float* oe = out_edge + (size_t)e * 128;
    float* ag = agg + (size_t)recvs[row] * 128;
#pragma unroll
    for (int q = 0; q < 32; ++q) {
      const int c = s4 * 32 + q;
      const float ne = gam[c] * (H2[row * 128 + c] - mu) * inv + bet[c];
      oe[c] = ne + ef[c];        // residual output
      atomicAdd(&ag[c], ne);     // fused scatter-add (L2-resident target)
    }
  }
}

// ---------------------------------------------------------------------------
// Node update: new_node = LN(relu([node|agg] @ W1 + b1) @ W2 + b2) + node
// Block: 64 nodes, 256 threads (8 waves).
// ---------------------------------------------------------------------------
__global__ __launch_bounds__(256) void node_update_kernel(
    const float* __restrict__ node_feats, const float* __restrict__ agg,
    const __bf16* __restrict__ w1p, const float* __restrict__ b1,
    const __bf16* __restrict__ w2p, const float* __restrict__ b2,
    const float* __restrict__ gam, const float* __restrict__ bet,
    float* __restrict__ out_node, int nN) {
  // Region 0 (32 KB): bf16 input 64x256 during GEMM1; f32 64x128 after GEMM2.
  __shared__ __attribute__((aligned(32))) char smem[32768 + 16384];
  __bf16* Abuf = (__bf16*)smem;
  float*  H2   = (float*)smem;
  __bf16* Hbuf = (__bf16*)(smem + 32768);

  const int t   = threadIdx.x;
  const int row = t >> 2;
  const int s4  = t & 3;
  const int n   = blockIdx.x * 64 + row;
  const int nC  = (n < nN) ? n : (nN - 1);

  {
    const float* segs[2] = { node_feats + (size_t)nC * 128,
                             agg        + (size_t)nC * 128 };
#pragma unroll
    for (int sg = 0; sg < 2; ++sg) {
      const float4* sp = (const float4*)segs[sg] + s4 * 8;
      __bf16* dp = &Abuf[row * 256 + sg * 128 + s4 * 32];
#pragma unroll
      for (int q = 0; q < 8; ++q) {
        float4 v = sp[q];
        dp[q * 4 + 0] = (__bf16)v.x; dp[q * 4 + 1] = (__bf16)v.y;
        dp[q * 4 + 2] = (__bf16)v.z; dp[q * 4 + 3] = (__bf16)v.w;
      }
    }
  }
  __syncthreads();

  const int lane = t & 31;
  const int nt   = t >> 5;
  const int col  = nt * 16 + (lane & 15);
  const int rb   = (lane & 16) ? 8 : 0;

  v8f acc[4] = {};
#pragma unroll
  for (int kt = 0; kt < 8; ++kt) {   // K = 256
    v16bf b = load_b_frag(w1p, kt, nt, lane);
#pragma unroll
    for (int mt = 0; mt < 4; ++mt) {
      v16bf a = load_a_frag(Abuf + mt * 16 * 256, 256, lane, kt * 32);
      acc[mt] = WMMA_BF16(a, b, acc[mt]);
    }
  }
  {
    const float bb = b1[col];
#pragma unroll
    for (int mt = 0; mt < 4; ++mt)
#pragma unroll
      for (int v = 0; v < 8; ++v) {
        float h = acc[mt][v] + bb; h = h > 0.f ? h : 0.f;
        Hbuf[(mt * 16 + v + rb) * 128 + col] = (__bf16)h;
      }
  }
  __syncthreads();

  v8f d[4] = {};
#pragma unroll
  for (int kt = 0; kt < 4; ++kt) {   // K = 128
    v16bf b = load_b_frag(w2p, kt, nt, lane);
#pragma unroll
    for (int mt = 0; mt < 4; ++mt) {
      v16bf a = load_a_frag(Hbuf + mt * 16 * 128, 128, lane, kt * 32);
      d[mt] = WMMA_BF16(a, b, d[mt]);
    }
  }
  {
    const float bb = b2[col];
#pragma unroll
    for (int mt = 0; mt < 4; ++mt)
#pragma unroll
      for (int v = 0; v < 8; ++v)
        H2[(mt * 16 + v + rb) * 128 + col] = d[mt][v] + bb;
  }
  __syncthreads();

  float sum = 0.f, sq = 0.f;
#pragma unroll
  for (int q = 0; q < 32; ++q) {
    const float x = H2[row * 128 + s4 * 32 + q];
    sum += x; sq += x * x;
  }
#pragma unroll
  for (int m = 2; m >= 1; m >>= 1) {
    sum += __shfl_xor(sum, m);
    sq  += __shfl_xor(sq, m);
  }
  const float mu  = sum * (1.f / 128.f);
  const float var = sq * (1.f / 128.f) - mu * mu;
  const float inv = rsqrtf(var + 1e-5f);

  if (n < nN) {
    const float* nf = node_feats + (size_t)n * 128;
    float* on = out_node + (size_t)n * 128;
#pragma unroll
    for (int q = 0; q < 32; ++q) {
      const int c = s4 * 32 + q;
      const float nn = gam[c] * (H2[row * 128 + c] - mu) * inv + bet[c];
      on[c] = nn + nf[c];
    }
  }
}

// ---------------------------------------------------------------------------
// Launch
// ---------------------------------------------------------------------------
extern "C" void kernel_launch(void* const* d_in, const int* in_sizes, int n_in,
                              void* d_out, int out_size, void* d_ws, size_t ws_size,
                              hipStream_t stream) {
  const float* node_feats = (const float*)d_in[0];
  const float* edge_feats = (const float*)d_in[1];
  const int*   senders    = (const int*)d_in[2];
  const int*   receivers  = (const int*)d_in[3];
  const float* ew1 = (const float*)d_in[4];  const float* eb1 = (const float*)d_in[5];
  const float* ew2 = (const float*)d_in[6];  const float* eb2 = (const float*)d_in[7];
  const float* eg  = (const float*)d_in[8];  const float* eo  = (const float*)d_in[9];
  const float* nw1 = (const float*)d_in[10]; const float* nb1 = (const float*)d_in[11];
  const float* nw2 = (const float*)d_in[12]; const float* nb2 = (const float*)d_in[13];
  const float* ng  = (const float*)d_in[14]; const float* no_ = (const float*)d_in[15];

  const int nNodes = in_sizes[0] / 128;
  const int nEdges = in_sizes[1] / 128;

  float* out_node = (float*)d_out;
  float* out_edge = out_node + (size_t)nNodes * 128;

  // Workspace layout: agg (f32, nNodes*128) then bf16-packed weights.
  char* ws = (char*)d_ws;
  size_t off = 0;
  float* agg = (float*)(ws + off);
  off += (size_t)nNodes * 128 * sizeof(float);
  off = (off + 255) & ~(size_t)255;
  __bf16* ew1p = (__bf16*)(ws + off); off += (size_t)384 * 128 * 2; off = (off + 255) & ~(size_t)255;
  __bf16* ew2p = (__bf16*)(ws + off); off += (size_t)128 * 128 * 2; off = (off + 255) & ~(size_t)255;
  __bf16* nw1p = (__bf16*)(ws + off); off += (size_t)256 * 128 * 2; off = (off + 255) & ~(size_t)255;
  __bf16* nw2p = (__bf16*)(ws + off); off += (size_t)128 * 128 * 2;

  hipMemsetAsync(agg, 0, (size_t)nNodes * 128 * sizeof(float), stream);

  pack_weights_kernel<<<(384 * 128 + 255) / 256, 256, 0, stream>>>(ew1, ew1p, 384);
  pack_weights_kernel<<<(128 * 128 + 255) / 256, 256, 0, stream>>>(ew2, ew2p, 128);
  pack_weights_kernel<<<(256 * 128 + 255) / 256, 256, 0, stream>>>(nw1, nw1p, 256);
  pack_weights_kernel<<<(128 * 128 + 255) / 256, 256, 0, stream>>>(nw2, nw2p, 128);

  edge_update_kernel<<<(nEdges + 63) / 64, 256, 0, stream>>>(
      node_feats, edge_feats, senders, receivers,
      ew1p, eb1, ew2p, eb2, eg, eo, agg, out_edge, nEdges);

  node_update_kernel<<<(nNodes + 63) / 64, 256, 0, stream>>>(
      node_feats, agg, nw1p, nb1, nw2p, nb2, ng, no_, out_node, nNodes);
}
